// FeedForwardNetwork_38336878084773
// MI455X (gfx1250) — matrix operand order repaired and checked by reading, not compiled, not run on hardware
//
#include <hip/hip_runtime.h>
#include <hip/hip_fp16.h>

typedef __attribute__((ext_vector_type(16))) _Float16 v16h;
typedef __attribute__((ext_vector_type(8)))  _Float16 v8h;
typedef __attribute__((ext_vector_type(8)))  float    v8f;

// ---------------------------------------------------------------------------
// Weight packing: (K x Nout) row-major f32  ->  WMMA-B-native f16 tiles.
// Tile (kt, nt) covers K[kt*32 .. +32) x N[nt*16 .. +16), stored at
// tileIdx = nt*KT + kt, 512 halves per tile. Within a tile, lane
// l = (k%32)/16*16 + n%16 holds halves h = k%16 (16 consecutive K values of
// column n). This contiguous k order matches the A fragments loaded in mma_job, so a
// B fragment is just 32 contiguous bytes per lane.
// ---------------------------------------------------------------------------
__global__ __launch_bounds__(256) void pack_w(const float* __restrict__ src,
                                              _Float16* __restrict__ dst,
                                              int K, int Nout) {
  int total = K * Nout;
  int KT = K >> 5;
  for (int tid = blockIdx.x * blockDim.x + threadIdx.x; tid < total;
       tid += gridDim.x * blockDim.x) {
    int k = tid / Nout;
    int n = tid - k * Nout;
    int ktile = k >> 5, kin = k & 31;
    int lane = ((kin >> 4) << 4) + (n & 15);
    int h = kin & 15;
    int ntile = n >> 4;
    long off = (((long)(ntile * KT + ktile)) << 9) + (lane << 4) + h;
    dst[off] = (_Float16)src[tid];
  }
}

// ---------------------------------------------------------------------------
// Core: one wave computes a 64-row x 16-col output slab (4 accumulators)
// for column-tile `ntile`. KT is a RUNTIME argument (host-supplied) so the
// compiler cannot fully unroll the K loop and hoist the whole LDS A tile
// into VGPRs (which caused scratch spills in the previous revision).
// A fragment per v_wmma_f32_16x16x32_f16: lane<16 -> row=lane, K k0+0..15;
// lane>=16 -> row=lane-16, K k0+16..31 (matches packed B).
// ---------------------------------------------------------------------------
template <int LDH>
__device__ __forceinline__ void mma_job(const _Float16* ldsA,
                                        const _Float16* __restrict__ wp,
                                        int ntile, int KT, int lane,
                                        v8f acc[4]) {
  v8f z = {};
  acc[0] = z; acc[1] = z; acc[2] = z; acc[3] = z;
  const int llo = lane & 15;
  const int koff = (lane >> 4) << 4;  // 0 or 16 halves
  const _Float16* bp = wp + (((long)(ntile * KT)) << 9) + (lane << 4);
  const _Float16* ap0 = ldsA + llo * LDH + koff;
#pragma clang loop unroll_count(2)
  for (int kt = 0; kt < KT; ++kt) {
    v16h b = *(const v16h*)bp;
    bp += 512;
    const _Float16* ap = ap0 + kt * 32;
#pragma unroll
    for (int ms = 0; ms < 4; ++ms) {
      v8h lo = *(const v8h*)(ap);
      v8h hi = *(const v8h*)(ap + 8);
      ap += 16 * LDH;
      v16h a = __builtin_shufflevector(lo, hi, 0, 1, 2, 3, 4, 5, 6, 7, 8, 9,
                                       10, 11, 12, 13, 14, 15);
      acc[ms] = __builtin_amdgcn_wmma_f32_16x16x32_f16(
          false, a, false, b, (short)0, acc[ms], false, false);
    }
  }
}

__device__ __forceinline__ float sigmoidf_(float y) {
  return 1.0f / (1.0f + __expf(-y));
}

// ---------------------------------------------------------------------------
// Stage 1, scalar irreps: mid0 = s @ w1_0 * attr/sqrt(512) + b1
//   cols [0,1024):  scal = silu(mid0)          -> f16 ws
//   cols [1024,1792): gates = sigmoid(mid0)    -> f16 ws
// ---------------------------------------------------------------------------
__global__ __launch_bounds__(256) void k1_scalar(
    const float* __restrict__ in, const float* __restrict__ attr,
    const _Float16* __restrict__ w, const float* __restrict__ b1,
    _Float16* __restrict__ scal, _Float16* __restrict__ gates, int N,
    int KT) {
  __shared__ __align__(16) _Float16 A[64 * 520];
  __shared__ float sattr[64];
  const long row0 = (long)blockIdx.x * 64;
  const int tid = threadIdx.x;
  for (int idx = tid; idx < 64 * 128; idx += 256) {
    int r = idx >> 7, c4 = (idx & 127) << 2;
    long gm = row0 + r;
    float4 v = make_float4(0.f, 0.f, 0.f, 0.f);
    if (gm < N) v = *(const float4*)(in + gm * 1920 + c4);
    _Float16* d = A + r * 520 + c4;
    d[0] = (_Float16)v.x; d[1] = (_Float16)v.y;
    d[2] = (_Float16)v.z; d[3] = (_Float16)v.w;
  }
  if (tid < 64) {
    long gm = row0 + tid;
    sattr[tid] = (gm < N) ? attr[gm] : 0.f;
  }
  __syncthreads();
  const int wv = tid >> 5, lane = tid & 31;
  const int llo = lane & 15, lhi = lane >> 4;
  for (int nt = wv; nt < 112; nt += 8) {
    v8f acc[4];
    mma_job<520>(A, w, nt, KT, lane, acc);
    int n = nt * 16 + llo;
    float bias = b1[n];
#pragma unroll
    for (int ms = 0; ms < 4; ++ms) {
#pragma unroll
      for (int r = 0; r < 8; ++r) {
        int rit = ms * 16 + lhi * 8 + r;
        long gm = row0 + rit;
        if (gm < N) {
          float y = acc[ms][r] * (sattr[rit] * 0.044194173824159216f) + bias;
          float sg = sigmoidf_(y);
          if (n < 1024) scal[gm * 1024 + n] = (_Float16)(y * sg);
          else          gates[gm * 768 + (n - 1024)] = (_Float16)sg;
        }
      }
    }
  }
}

// ---------------------------------------------------------------------------
// Stage 1, vector irreps (3 components): h1[m] = (v[:,:,m] @ w1_1) *
//   attr/16 * gates[:, n]   -> f16 planes [3][N][512]
// ---------------------------------------------------------------------------
__global__ __launch_bounds__(256) void k2_vec(
    const float* __restrict__ in, const float* __restrict__ attr,
    const _Float16* __restrict__ w, const _Float16* __restrict__ gates,
    _Float16* __restrict__ h1, int N, int KT) {
  __shared__ __align__(16) _Float16 A[3 * 64 * 264];
  __shared__ float sattr[64];
  const long row0 = (long)blockIdx.x * 64;
  const int tid = threadIdx.x;
  for (int idx = tid; idx < 64 * 768; idx += 256) {
    int r = idx / 768, j = idx - r * 768;
    long gm = row0 + r;
    float v = (gm < N) ? in[gm * 1920 + 512 + j] : 0.f;
    int comp = j % 3, i = j / 3;
    A[comp * (64 * 264) + r * 264 + i] = (_Float16)v;
  }
  if (tid < 64) {
    long gm = row0 + tid;
    sattr[tid] = (gm < N) ? attr[gm] : 0.f;
  }
  __syncthreads();
  const int wv = tid >> 5, lane = tid & 31;
  const int llo = lane & 15, lhi = lane >> 4;
  for (int jb = wv; jb < 96; jb += 8) {
    int comp = jb % 3, nt = jb / 3;
    v8f acc[4];
    mma_job<264>(A + comp * (64 * 264), w, nt, KT, lane, acc);
    int n = nt * 16 + llo;
#pragma unroll
    for (int ms = 0; ms < 4; ++ms) {
#pragma unroll
      for (int r = 0; r < 8; ++r) {
        int rit = ms * 16 + lhi * 8 + r;
        long gm = row0 + rit;
        if (gm < N) {
          float g = (float)gates[gm * 768 + n];
          float val = acc[ms][r] * (sattr[rit] * 0.0625f) * g;
          h1[(long)comp * N * 512 + gm * 512 + n] = (_Float16)val;
        }
      }
    }
  }
}

// ---------------------------------------------------------------------------
// Stage 1, tensor irreps (5 components): h2[m] = (t[:,:,m] @ w1_2) *
//   attr/sqrt(128) * gates[:, 512+n]   -> f16 planes [5][N][256]
// ---------------------------------------------------------------------------
__global__ __launch_bounds__(256) void k3_tens(
    const float* __restrict__ in, const float* __restrict__ attr,
    const _Float16* __restrict__ w, const _Float16* __restrict__ gates,
    _Float16* __restrict__ h2, int N, int KT) {
  __shared__ __align__(16) _Float16 A[5 * 64 * 136];
  __shared__ float sattr[64];
  const long row0 = (long)blockIdx.x * 64;
  const int tid = threadIdx.x;
  for (int idx = tid; idx < 64 * 640; idx += 256) {
    int r = idx / 640, j = idx - r * 640;
    long gm = row0 + r;
    float v = (gm < N) ? in[gm * 1920 + 1280 + j] : 0.f;
    int comp = j % 5, i = j / 5;
    A[comp * (64 * 136) + r * 136 + i] = (_Float16)v;
  }
  if (tid < 64) {
    long gm = row0 + tid;
    sattr[tid] = (gm < N) ? attr[gm] : 0.f;
  }
  __syncthreads();
  const int wv = tid >> 5, lane = tid & 31;
  const int llo = lane & 15, lhi = lane >> 4;
  for (int jb = wv; jb < 80; jb += 8) {
    int comp = jb % 5, nt = jb / 5;
    v8f acc[4];
    mma_job<136>(A + comp * (64 * 136), w, nt, KT, lane, acc);
    int n = nt * 16 + llo;
#pragma unroll
    for (int ms = 0; ms < 4; ++ms) {
#pragma unroll
      for (int r = 0; r < 8; ++r) {
        int rit = ms * 16 + lhi * 8 + r;
        long gm = row0 + rit;
        if (gm < N) {
          float g = (float)gates[gm * 768 + 512 + n];
          float val = acc[ms][r] * (sattr[rit] * 0.08838834764831845f) * g;
          h2[(long)comp * N * 256 + gm * 256 + n] = (_Float16)val;
        }
      }
    }
  }
}

// ---------------------------------------------------------------------------
// Stage 2, scalar: out0 = scal @ w2_0 * attr/32 + b2  -> d_out cols [0,512)
// ---------------------------------------------------------------------------
__global__ __launch_bounds__(256) void k4_scalar(
    const _Float16* __restrict__ scal, const float* __restrict__ attr,
    const _Float16* __restrict__ w, const float* __restrict__ b2,
    float* __restrict__ out, int N, int KT) {
  __shared__ __align__(16) _Float16 A[64 * 1032];
  __shared__ float sattr[64];
  const long row0 = (long)blockIdx.x * 64;
  const int tid = threadIdx.x;
  for (int idx = tid; idx < 64 * 128; idx += 256) {
    int r = idx >> 7, c8 = (idx & 127) << 3;
    long gm = row0 + r;
    v8h v = {};
    if (gm < N) v = *(const v8h*)(scal + gm * 1024 + c8);
    *(v8h*)(A + r * 1032 + c8) = v;
  }
  if (tid < 64) {
    long gm = row0 + tid;
    sattr[tid] = (gm < N) ? attr[gm] : 0.f;
  }
  __syncthreads();
  const int wv = tid >> 5, lane = tid & 31;
  const int llo = lane & 15, lhi = lane >> 4;
  for (int nt = wv; nt < 32; nt += 8) {
    v8f acc[4];
    mma_job<1032>(A, w, nt, KT, lane, acc);
    int n = nt * 16 + llo;
    float bias = b2[n];
#pragma unroll
    for (int ms = 0; ms < 4; ++ms) {
#pragma unroll
      for (int r = 0; r < 8; ++r) {
        int rit = ms * 16 + lhi * 8 + r;
        long gm = row0 + rit;
        if (gm < N)
          out[gm * 1920 + n] = acc[ms][r] * (sattr[rit] * 0.03125f) + bias;
      }
    }
  }
}

// ---------------------------------------------------------------------------
// Stage 2, vector: out1[:,n,m] = (h1[m] @ w2_1) * attr/sqrt(512)
//   -> d_out cols 512 + n*3 + m. grid.y = component m.
// ---------------------------------------------------------------------------
__global__ __launch_bounds__(256) void k5_vec(
    const _Float16* __restrict__ h1, const float* __restrict__ attr,
    const _Float16* __restrict__ w, float* __restrict__ out, int N, int KT) {
  __shared__ __align__(16) _Float16 A[64 * 520];
  __shared__ float sattr[64];
  const long row0 = (long)blockIdx.x * 64;
  const int m = blockIdx.y;
  const _Float16* src = h1 + (long)m * N * 512;
  const int tid = threadIdx.x;
  for (int idx = tid; idx < 64 * 64; idx += 256) {
    int r = idx >> 6, c8 = (idx & 63) << 3;
    long gm = row0 + r;
    v8h v = {};
    if (gm < N) v = *(const v8h*)(src + gm * 512 + c8);
    *(v8h*)(A + r * 520 + c8) = v;
  }
  if (tid < 64) {
    long gm = row0 + tid;
    sattr[tid] = (gm < N) ? attr[gm] : 0.f;
  }
  __syncthreads();
  const int wv = tid >> 5, lane = tid & 31;
  const int llo = lane & 15, lhi = lane >> 4;
  for (int nt = wv; nt < 16; nt += 8) {
    v8f acc[4];
    mma_job<520>(A, w, nt, KT, lane, acc);
    int n = nt * 16 + llo;
#pragma unroll
    for (int ms = 0; ms < 4; ++ms) {
#pragma unroll
      for (int r = 0; r < 8; ++r) {
        int rit = ms * 16 + lhi * 8 + r;
        long gm = row0 + rit;
        if (gm < N)
          out[gm * 1920 + 512 + n * 3 + m] =
              acc[ms][r] * (sattr[rit] * 0.044194173824159216f);
      }
    }
  }
}

// ---------------------------------------------------------------------------
// Stage 2, tensor: out2[:,n,m] = (h2[m] @ w2_2) * attr/16
//   -> d_out cols 1280 + n*5 + m. grid.y = component m.
// ---------------------------------------------------------------------------
__global__ __launch_bounds__(256) void k6_tens(
    const _Float16* __restrict__ h2, const float* __restrict__ attr,
    const _Float16* __restrict__ w, float* __restrict__ out, int N, int KT) {
  __shared__ __align__(16) _Float16 A[64 * 264];
  __shared__ float sattr[64];
  const long row0 = (long)blockIdx.x * 64;
  const int m = blockIdx.y;
  const _Float16* src = h2 + (long)m * N * 256;
  const int tid = threadIdx.x;
  for (int idx = tid; idx < 64 * 32; idx += 256) {
    int r = idx >> 5, c8 = (idx & 31) << 3;
    long gm = row0 + r;
    v8h v = {};
    if (gm < N) v = *(const v8h*)(src + gm * 256 + c8);
    *(v8h*)(A + r * 264 + c8) = v;
  }
  if (tid < 64) {
    long gm = row0 + tid;
    sattr[tid] = (gm < N) ? attr[gm] : 0.f;
  }
  __syncthreads();
  const int wv = tid >> 5, lane = tid & 31;
  const int llo = lane & 15, lhi = lane >> 4;
  for (int nt = wv; nt < 8; nt += 8) {
    v8f acc[4];
    mma_job<264>(A, w, nt, KT, lane, acc);
    int n = nt * 16 + llo;
#pragma unroll
    for (int ms = 0; ms < 4; ++ms) {
#pragma unroll
      for (int r = 0; r < 8; ++r) {
        int rit = ms * 16 + lhi * 8 + r;
        long gm = row0 + rit;
        if (gm < N)
          out[gm * 1920 + 1280 + n * 5 + m] =
              acc[ms][r] * (sattr[rit] * 0.0625f);
      }
    }
  }
}

extern "C" void kernel_launch(void* const* d_in, const int* in_sizes, int n_in,
                              void* d_out, int out_size, void* d_ws,
                              size_t ws_size, hipStream_t stream) {
  const float* node_input = (const float*)d_in[0];
  const float* node_attr  = (const float*)d_in[1];
  const float* w1_0 = (const float*)d_in[2];
  const float* w1_1 = (const float*)d_in[3];
  const float* w1_2 = (const float*)d_in[4];
  const float* b1   = (const float*)d_in[5];
  const float* w2_0 = (const float*)d_in[6];
  const float* w2_1 = (const float*)d_in[7];
  const float* w2_2 = (const float*)d_in[8];
  const float* b2   = (const float*)d_in[9];
  float* out = (float*)d_out;
  const int N = in_sizes[0] / 1920;

  _Float16* ws = (_Float16*)d_ws;
  _Float16* w1_0p = ws;                       // 512*1792 = 917504 halves
  _Float16* w1_1p = w1_0p + 512 * 1792;       // 131072
  _Float16* w1_2p = w1_1p + 256 * 512;        // 32768
  _Float16* w2_0p = w1_2p + 128 * 256;        // 524288
  _Float16* w2_1p = w2_0p + 1024 * 512;       // 131072
  _Float16* w2_2p = w2_1p + 512 * 256;        // 32768
  _Float16* scal  = w2_2p + 256 * 128;        // N*1024
  _Float16* gates = scal + (long)N * 1024;    // N*768
  _Float16* h1    = gates + (long)N * 768;    // 3*N*512
  _Float16* h2    = h1 + 3L * N * 512;        // 5*N*256

  // Weight packing (small, one-time per launch).
  pack_w<<<dim3((512 * 1792 + 255) / 256), 256, 0, stream>>>(w1_0, w1_0p, 512, 1792);
  pack_w<<<dim3((256 * 512 + 255) / 256), 256, 0, stream>>>(w1_1, w1_1p, 256, 512);
  pack_w<<<dim3((128 * 256 + 255) / 256), 256, 0, stream>>>(w1_2, w1_2p, 128, 256);
  pack_w<<<dim3((1024 * 512 + 255) / 256), 256, 0, stream>>>(w2_0, w2_0p, 1024, 512);
  pack_w<<<dim3((512 * 256 + 255) / 256), 256, 0, stream>>>(w2_1, w2_1p, 512, 256);
  pack_w<<<dim3((256 * 128 + 255) / 256), 256, 0, stream>>>(w2_2, w2_2p, 256, 128);

  const int RT = (N + 63) / 64;
  // KT (K/32) passed as a runtime value to keep the K loops rolled.
  k1_scalar<<<dim3(RT), 256, 0, stream>>>(node_input, node_attr, w1_0p, b1, scal, gates, N, 512 / 32);
  k2_vec   <<<dim3(RT), 256, 0, stream>>>(node_input, node_attr, w1_1p, gates, h1, N, 256 / 32);
  k3_tens  <<<dim3(RT), 256, 0, stream>>>(node_input, node_attr, w1_2p, gates, h2, N, 128 / 32);
  k4_scalar<<<dim3(RT), 256, 0, stream>>>(scal, node_attr, w2_0p, b2, out, N, 1024 / 32);
  k5_vec   <<<dim3(RT, 3), 256, 0, stream>>>(h1, node_attr, w2_1p, out, N, 512 / 32);
  k6_tens  <<<dim3(RT, 5), 256, 0, stream>>>(h2, node_attr, w2_2p, out, N, 256 / 32);
}
